// GenericRSPlaqCouplingLayer_18519898980818
// MI455X (gfx1250) — matrix operand order repaired and checked by reading, not compiled, not run on hardware
//
#include <hip/hip_runtime.h>
#include <math.h>

// ---------------------------------------------------------------------------
// Types
// ---------------------------------------------------------------------------
typedef _Float16 v16h __attribute__((ext_vector_type(16)));
typedef _Float16 v8h  __attribute__((ext_vector_type(8)));
typedef float    v8f  __attribute__((ext_vector_type(8)));
typedef unsigned int u32x4 __attribute__((ext_vector_type(4)));
typedef int i32x4 __attribute__((ext_vector_type(4)));
typedef int i32x8 __attribute__((ext_vector_type(8)));

static constexpr int   LDIM   = 256;
static constexpr float TWO_PI_F = 6.283185307179586476925f;

// ---------------------------------------------------------------------------
// Helpers
// ---------------------------------------------------------------------------
static __device__ inline v16h make16(v8h lo, v8h hi) {
  v16h r;
#pragma unroll
  for (int i = 0; i < 8; ++i) { r[i] = lo[i]; r[i + 8] = hi[i]; }
  return r;
}

static __device__ inline v8f wmma_f16f32(v16h a, v16h b, v8f c) {
  // D = A(16x32 f16) * B(32x16 f16) + C(16x16 f32)
  return __builtin_amdgcn_wmma_f32_16x16x32_f16(false, a, false, b, (short)0, c,
                                                false, false);
}

static __device__ inline float gelu_tanh(float v) {
  return 0.5f * v * (1.0f + tanhf(0.7978845608028654f * (v + 0.044715f * v * v * v)));
}

// ---------------------------------------------------------------------------
// TDM: 1-row tensor load (contiguous n8*8 bytes) global -> LDS, per ISA D#
// group0: [1:0]=count=1, [63:32]=lds_addr, [120:64]=global_addr, [127:126]=type=2
// group1: data_size=3 (8B), tensor_dim0=tile_dim0=n8, tensor_dim1=tile_dim1=1
// ---------------------------------------------------------------------------
#if __has_builtin(__builtin_amdgcn_tensor_load_to_lds)
#define HAVE_TDM 1
static __device__ inline void tdm_row_load(const void* gsrc, unsigned lds_off,
                                           unsigned n8) {
  unsigned long long ga = (unsigned long long)(uintptr_t)gsrc;
  u32x4 g0 = { 1u, lds_off, (unsigned)ga,
               ((unsigned)(ga >> 32) & 0x01FFFFFFu) | 0x80000000u };
  i32x8 g1 = { (int)(3u << 16),                          // data_size = 8B
               (int)((n8 & 0xFFFFu) << 16),              // tensor_dim0 lo
               (int)((1u << 16) | (n8 >> 16)),           // tensor_dim1=1 | dim0 hi
               (int)((n8 & 0xFFFFu) << 16),              // tile_dim0
               1,                                        // tile_dim1=1, tile_dim2=0
               (int)n8,                                  // tensor_dim0_stride lo
               0, 0 };
  i32x4 z4 = {0, 0, 0, 0};
#if defined(__clang_major__) && (__clang_major__ >= 23)
  i32x8 z8 = {0, 0, 0, 0, 0, 0, 0, 0};
  __builtin_amdgcn_tensor_load_to_lds(g0, g1, z4, z4, z8, 0);
#else
  __builtin_amdgcn_tensor_load_to_lds(g0, g1, z4, z4, 0);
#endif
}
#endif

// ---------------------------------------------------------------------------
// Kernel 1: pre-swizzle conv weights (f32 -> f16) into per-lane WMMA B-fragment
// order so each lane's 16-half B fragment is one contiguous 32B LDS load.
// B-fragment layout (ISA 7.12.2): lane l -> n = l%16 ; half-slot s -> k = (s&7) + 8*(l/16) + (s>=8 ? 16 : 0)
// conv1 K order: k = tap*2 + c      (K=18, padded to 32)
// conv2 K order: k = tap*32 + c     (K=288 = 9 chunks of 32)
// ---------------------------------------------------------------------------
__global__ __launch_bounds__(256) void prep_weights(
    const float* __restrict__ c1w,   // (32, 2, 3, 3)
    const float* __restrict__ c2w,   // (25, 32, 3, 3)
    _Float16* __restrict__ w1o,      // [2 ntile][32 lane][16 slot]   = 1024
    _Float16* __restrict__ w2o) {    // [9 kc][2 ntile][32 lane][16]  = 9216
  int e = blockIdx.x * 256 + threadIdx.x;
  if (e >= 10240) return;
  if (e < 1024) {
    int s = e & 15, lane = (e >> 4) & 31, ntile = e >> 9;
    int k = (s & 7) + 8 * (lane >> 4) + ((s & 8) ? 16 : 0);
    int n = ntile * 16 + (lane & 15);
    float v = 0.0f;
    if (k < 18) {
      int tap = k >> 1, c = k & 1;               // tap = ky*3+kx
      v = c1w[n * 18 + c * 9 + tap];
    }
    w1o[e] = (_Float16)v;
  } else {
    int e2 = e - 1024;
    int s = e2 & 15, lane = (e2 >> 4) & 31;
    int nt = (e2 >> 9) & 1, kc = e2 >> 10;       // kc = tap
    int c = (s & 7) + 8 * (lane >> 4) + ((s & 8) ? 16 : 0);  // input channel 0..31
    int n = nt * 16 + (lane & 15);
    float v = (n < 25) ? c2w[n * 288 + c * 9 + kc] : 0.0f;
    w2o[e2] = (_Float16)v;
  }
}

// ---------------------------------------------------------------------------
// Kernel 2: fully fused tile kernel. One block = one 16x16 output tile.
// LDS layout (single static block, assumed at LDS offset 0 for TDM):
//   [0      ) w1f  : 1024 f16  (2048 B)     -- reused as f32 red[256] at end
//   [2048   ) w2f  : 9216 f16  (18432 B)
//   [20480  ) nin  : 400*2 f16 (1600 B, pad 1664)   20x20 halo-2 cos/sin
//   [22144  ) hbuf : 324*32 f16 (20736 B)           18x18 halo-1 hidden
//   [42880  ) nout : 256*26 f16 (13312 B)           16x16 x 25(+pad) logits
//   total 56192 B  ( < 64 KB, far under the 320 KB WGP budget )
// ---------------------------------------------------------------------------
__global__ __launch_bounds__(256) void fused_spline(
    const float* __restrict__ x, const float* __restrict__ act,
    const float* __restrict__ froz, const float* __restrict__ pas,
    const float* __restrict__ c1b, const float* __restrict__ c2b,
    const _Float16* __restrict__ w1g, const _Float16* __restrict__ w2g,
    float* __restrict__ fxout, float* __restrict__ partial) {
  __shared__ __align__(16) char SMEM[56192];
  _Float16* w1f  = (_Float16*)(SMEM);
  _Float16* w2f  = (_Float16*)(SMEM + 2048);
  _Float16* nin  = (_Float16*)(SMEM + 20480);
  _Float16* hbuf = (_Float16*)(SMEM + 22144);
  _Float16* nout = (_Float16*)(SMEM + 42880);
  float* red = (float*)SMEM;  // reuse after conv2

  const int t = threadIdx.x;
  const int lane = t & 31, wv = t >> 5;
  const int grp = lane >> 4, ln16 = lane & 15;
  const int tile = blockIdx.x;
  const int b = tile >> 8, ty = (tile >> 4) & 15, tx = tile & 15;
  const int gy0 = ty * 16, gx0 = tx * 16;
  const float* xb = x + (size_t)b * (LDIM * LDIM);

#ifdef HAVE_TDM
  if (wv == 0) {  // one wave DMAs the swizzled weights into LDS
    tdm_row_load(w1g, 0u, 256u);      //  2048 B -> lds 0
    tdm_row_load(w2g, 2048u, 2304u);  // 18432 B -> lds 2048
#if __has_builtin(__builtin_amdgcn_s_wait_tensorcnt)
    __builtin_amdgcn_s_wait_tensorcnt(0);
#endif
  }
#endif

  // net_in over the 20x20 halo-2 region (circular wrap; L is a power of two)
  for (int i = t; i < 400; i += 256) {
    int yy = i / 20, xx = i % 20;
    int gy = (gy0 - 2 + yy) & (LDIM - 1);
    int gx = (gx0 - 2 + xx) & (LDIM - 1);
    float p2 = froz[gy * LDIM + gx] * xb[gy * LDIM + gx];
    nin[i * 2 + 0] = (_Float16)cosf(p2);
    nin[i * 2 + 1] = (_Float16)sinf(p2);
  }
  // Authoritative weight copy (same bytes the TDM fetched).
  for (int i = t; i < 1024; i += 256) w1f[i] = w1g[i];
  for (int i = t; i < 9216; i += 256) w2f[i] = w2g[i];
  __syncthreads();

  // ---- conv1: 2 -> 32 channels over 18x18 region (324 px, 21 M-tiles), GELU
  for (int job = wv; job < 42; job += 8) {
    int mtile = job % 21, ntile = job / 21;
    int ncol = ntile * 16 + ln16;
    int hpix = mtile * 16 + ln16;       // A-fragment row (m = lane%16)
    bool mv = hpix < 324;
    int hy = mv ? hpix / 18 : 0, hx = mv ? hpix % 18 : 0;
    v16h A;
#pragma unroll
    for (int s = 0; s < 16; ++s) {
      // A layout: k = s + 8*grp + (s>=8 ? 8 : 0)
      int k = s + 8 * grp + (s >= 8 ? 8 : 0);
      _Float16 v = (_Float16)0.0f;
      if (k < 18) {
        int tap = k >> 1, c = k & 1;
        int dy = tap / 3 - 1, dx = tap % 3 - 1;
        v = nin[((hy + 1 + dy) * 20 + (hx + 1 + dx)) * 2 + c];
      }
      A[s] = v;
    }
    const v8h* wp = (const v8h*)(w1f + (ntile * 32 + lane) * 16);
    v16h B = make16(wp[0], wp[1]);
    v8f acc = {};
    acc = wmma_f16f32(A, B, acc);
    float bias = c1b[ncol];
#pragma unroll
    for (int r = 0; r < 8; ++r) {       // D layout: m = 8*grp + r, n = lane%16
      int hp = mtile * 16 + 8 * grp + r;
      if (hp < 324) hbuf[hp * 32 + ncol] = (_Float16)gelu_tanh(acc[r] + bias);
    }
  }
  __syncthreads();

  // ---- conv2: 32 -> 25 channels, K = 9 taps x 32 ch. Each wave: 1 N-tile x 4 M-tiles.
  {
    int ntile = wv >> 2;
    int ncol = ntile * 16 + ln16;
    int mbase = (wv & 3) * 4;
    v8f acc[4] = {};
    for (int kc = 0; kc < 9; ++kc) {
      const v8h* wp = (const v8h*)(w2f + ((kc * 2 + ntile) * 32 + lane) * 16);
      v16h B = make16(wp[0], wp[1]);
      int dy = kc / 3 - 1, dx = kc % 3 - 1;
#pragma unroll
      for (int j = 0; j < 4; ++j) {
        int pix = (mbase + j) * 16 + ln16;            // A row
        int py = pix >> 4, px = pix & 15;
        const _Float16* hp = hbuf + ((py + 1 + dy) * 18 + (px + 1 + dx)) * 32;
        // per-lane A = channels {8g..8g+7, 16+8g..16+8g+7}: two ds_load_b128
        v16h A = make16(*(const v8h*)(hp + 8 * grp),
                        *(const v8h*)(hp + 16 + 8 * grp));
        acc[j] = wmma_f16f32(A, B, acc[j]);
      }
    }
    float bias = (ncol < 25) ? c2b[ncol] : 0.0f;
#pragma unroll
    for (int j = 0; j < 4; ++j)
#pragma unroll
      for (int r = 0; r < 8; ++r) {
        int pix = (mbase + j) * 16 + 8 * grp + r;
        if (ncol < 25) nout[pix * 26 + ncol] = (_Float16)(acc[j][r] + bias);
      }
  }
  __syncthreads();

  // ---- spline: one thread per pixel of the 16x16 tile (f32 VALU) ----
  {
    int py = t >> 4, px = t & 15;
    int gi = (gy0 + py) * LDIM + (gx0 + px);
    float no[25];
#pragma unroll
    for (int c = 0; c < 25; ++c) no[c] = (float)nout[t * 26 + c];

    float pw[8], ph[8], dd[8];
    { float mx = no[0];
      for (int c = 1; c < 8; ++c) mx = fmaxf(mx, no[c]);
      float sm = 0.f;
      for (int c = 0; c < 8; ++c) { pw[c] = expf(no[c] - mx); sm += pw[c]; }
      float inv = 1.0f / sm;
      for (int c = 0; c < 8; ++c) pw[c] *= inv; }
    { float mx = no[8];
      for (int c = 1; c < 8; ++c) mx = fmaxf(mx, no[8 + c]);
      float sm = 0.f;
      for (int c = 0; c < 8; ++c) { ph[c] = expf(no[8 + c] - mx); sm += ph[c]; }
      float inv = 1.0f / sm;
      for (int c = 0; c < 8; ++c) ph[c] *= inv; }
    for (int c = 0; c < 8; ++c) {
      float v = no[16 + c];
      dd[c] = (v > 20.0f) ? v : log1pf(expf(v));
    }
    float tt = no[24];

    float kx_[9], ky_[9];
    kx_[0] = 0.f; ky_[0] = 0.f;
    float cw = 0.f, chh = 0.f;
    for (int c = 0; c < 8; ++c) {
      cw += pw[c]; chh += ph[c];
      kx_[c + 1] = TWO_PI_F * cw; ky_[c + 1] = TWO_PI_F * chh;
    }
    kx_[8] = TWO_PI_F; ky_[8] = TWO_PI_F;

    float a = act[gi], f = froz[gi], p = pas[gi];
    float xv = xb[gi];
    float x1 = a * fmodf(xv, TWO_PI_F);
    int idx = 0;
    for (int c = 1; c < 8; ++c) idx += (kx_[c] <= x1) ? 1 : 0;
    if (idx > 7) idx = 7;
    float xk = kx_[idx], xk1 = kx_[idx + 1];
    float yk = ky_[idx], yk1 = ky_[idx + 1];
    float dk = dd[idx], dk1 = (idx + 1 < 8) ? dd[idx + 1] : dd[0];
    float wk = xk1 - xk, hk = yk1 - yk;
    float sl = hk / wk;
    float th = (x1 - xk) / wk;
    float th1 = th * (1.0f - th);
    float den = sl + (dk1 + dk - 2.0f * sl) * th1;
    float fx1 = yk + hk * (sl * th * th + dk * th1) / den;
    float omt = 1.0f - th;
    float deriv = sl * sl * (dk1 * th * th + 2.0f * sl * th1 + dk * omt * omt) /
                  (den * den);
    fx1 = a * fx1;
    float lj = (a != 0.0f) ? a * logf(deriv) : 0.0f;
    float r1 = fmodf(fx1 + tt, TWO_PI_F);
    if (r1 < 0.0f) r1 += TWO_PI_F;
    fxout[(size_t)b * (LDIM * LDIM) + gi] = a * r1 + p * xv + f * xv;
    red[t] = lj;
  }
  __syncthreads();
  for (int o = 128; o; o >>= 1) {
    if (t < o) red[t] += red[t + o];
    __syncthreads();
  }
  if (t == 0) partial[blockIdx.x] = red[0];
}

// ---------------------------------------------------------------------------
// Kernel 3: deterministic per-batch reduction of the 256 tile partials.
// ---------------------------------------------------------------------------
__global__ __launch_bounds__(256) void reduce_logj(const float* __restrict__ part,
                                                   float* __restrict__ logj) {
  __shared__ float red[256];
  int b = blockIdx.x, t = threadIdx.x;
  red[t] = part[b * 256 + t];
  __syncthreads();
  for (int o = 128; o; o >>= 1) {
    if (t < o) red[t] += red[t + o];
    __syncthreads();
  }
  if (t == 0) logj[b] = red[0];
}

// ---------------------------------------------------------------------------
extern "C" void kernel_launch(void* const* d_in, const int* in_sizes, int n_in,
                              void* d_out, int out_size, void* d_ws, size_t ws_size,
                              hipStream_t stream) {
  const float* x    = (const float*)d_in[0];
  const float* act  = (const float*)d_in[1];
  const float* froz = (const float*)d_in[2];
  const float* pas  = (const float*)d_in[3];
  const float* c1w  = (const float*)d_in[4];
  const float* c1b  = (const float*)d_in[5];
  const float* c2w  = (const float*)d_in[6];
  const float* c2b  = (const float*)d_in[7];
  float* out = (float*)d_out;

  const int B = in_sizes[0] / (LDIM * LDIM);          // 16
  _Float16* w1s = (_Float16*)d_ws;                    // 1024 f16
  _Float16* w2s = w1s + 1024;                         // 9216 f16
  float* partial = (float*)((char*)d_ws + 20480);     // B*256 f32

  prep_weights<<<40, 256, 0, stream>>>(c1w, c2w, w1s, w2s);
  fused_spline<<<B * 256, 256, 0, stream>>>(x, act, froz, pas, c1b, c2b, w1s, w2s,
                                            out, partial);
  reduce_logj<<<B, 256, 0, stream>>>(partial, out + (size_t)B * LDIM * LDIM);
}
